// Attention_10505490006679
// MI455X (gfx1250) — compile-verified, matrix operand order checked by reading, not run on hardware
//
#include <hip/hip_runtime.h>
#include <hip/hip_bf16.h>

// Attention: out = softmax((x Wq)(x Wk)^T / sqrt(D)) (x Wv)
// N = 8192, D = 1024, fp32 in/out. Internals f16, f32 WMMA accumulate.
// Workspace (>= 176 MB): Q,K,V f16 (16 MB each) + S/P f16 (128 MB, in-place softmax).

#define AN 8192
#define AD 1024

#define BM 128
#define BN 128
#define BK 32
#define LST 40   // LDS row stride in halves (80B = 20 banks -> conflict-free frag loads)

typedef _Float16 v16h __attribute__((ext_vector_type(16)));
typedef _Float16 v8h  __attribute__((ext_vector_type(8)));
typedef float    v8f  __attribute__((ext_vector_type(8)));
typedef int      v4i_ __attribute__((ext_vector_type(4)));

// ---- CDNA5 async global->LDS copy (ASYNCcnt), with safe fallback ----
#if defined(__has_builtin)
#  if __has_builtin(__builtin_amdgcn_global_load_async_to_lds_b128)
#    define USE_ASYNC_LDS 1
#  endif
#endif
#ifndef USE_ASYNC_LDS
#  define USE_ASYNC_LDS 0
#endif

#if USE_ASYNC_LDS
typedef __attribute__((address_space(1))) v4i_* g4p_t;   // global int4*
typedef __attribute__((address_space(3))) v4i_* l4p_t;   // LDS int4*
#endif

// Copy 16 bytes (8 halves) global -> LDS.
__device__ __forceinline__ void cp_b128(const _Float16* g, _Float16* l) {
#if USE_ASYNC_LDS
    // int->pointer casts sidestep addrspacecast restrictions; flat-shared
    // pointer low 32 bits == LDS byte offset (ISA flat addressing rule).
    __builtin_amdgcn_global_load_async_to_lds_b128(
        (g4p_t)(unsigned long long)g,
        (l4p_t)(unsigned)(unsigned long long)l,
        0, 0);
#else
    *(v8h*)l = *(const v8h*)g;
#endif
}

__device__ __forceinline__ void async_wait_all() {
#if USE_ASYNC_LDS
#  if __has_builtin(__builtin_amdgcn_s_wait_asynccnt)
    __builtin_amdgcn_s_wait_asynccnt(0);
#  else
    asm volatile("s_wait_asynccnt 0x0" ::: "memory");
#  endif
#endif
}

__device__ __forceinline__ v8f wmma_f16(v16h a, v16h b, v8f c) {
    return __builtin_amdgcn_wmma_f32_16x16x32_f16(
        false, a, false, b, (short)0, c, false, false);
}

// Per-lane WMMA fragment from LDS tile stored [row][k], row stride LST halves.
// slots 0..7 -> k = hi*8.., slots 8..15 -> k = hi*8+16..
__device__ __forceinline__ v16h load_frag(const _Float16* p) {
    v8h lo = *(const v8h*)(p);
    v8h hh = *(const v8h*)(p + 16);
    v16h r;
#pragma unroll
    for (int i = 0; i < 8; ++i) { r[i] = lo[i]; r[i + 8] = hh[i]; }
    return r;
}

// ======================================================================
// C[M,N] = alpha * A[M,Kd] * B[Kd,N]  (B row-major; transposed into LDS)
// Double-buffered pipeline. A-operand: async global->LDS when f16,
// register-staged f32->f16 conversion otherwise. B: register transpose.
// ======================================================================
template <typename TA, typename TB, typename TC>
__device__ __forceinline__ void gemm_bt(const TA* __restrict__ A, const TB* __restrict__ B,
                                        TC* __restrict__ C, int Kd, int Ncols, float alpha,
                                        int mBase, int nBase) {
    __shared__ __align__(16) _Float16 ldsA[2][BM * LST];
    __shared__ __align__(16) _Float16 ldsB[2][BM * LST];

    const int t    = threadIdx.x;
    const int lane = t & 31;
    const int wave = t >> 5;
    const int lr   = lane & 15;
    const int hi   = lane >> 4;
    const int waveM = wave & 3;   // 4 wave-rows (32 M each)
    const int waveN = wave >> 2;  // 2 wave-cols (64 N each)

    v8f acc[2][4];
#pragma unroll
    for (int mi = 0; mi < 2; ++mi)
#pragma unroll
        for (int ni = 0; ni < 4; ++ni) acc[mi][ni] = (v8f){0,0,0,0,0,0,0,0};

    const int arow = t >> 1;          // A loader: 2 threads/row, 16 elems each
    const int acol = (t & 1) * 16;
    const int bcol = t & 127;         // B loader: one out-column/thread, 16 k's
    const int bk0  = (t >> 7) * 16;

    const TA* aSrc = A + (size_t)(mBase + arow) * Kd + acol;
    const TB* bSrc = B + (size_t)(nBase + bcol);

    float areg[16];
    TB    breg[16];

    // Issue loads for K-step k0 targeting LDS buffer `buf`.
    auto fetch = [&](int k0, int buf) {
        if constexpr (sizeof(TA) == 2) {
            _Float16* da = &ldsA[buf][arow * LST + acol];
            cp_b128(aSrc + k0,     da);       // async straight into LDS
            cp_b128(aSrc + k0 + 8, da + 8);
        } else {
            *(float4*)(&areg[0])  = *(const float4*)(aSrc + k0);
            *(float4*)(&areg[4])  = *(const float4*)(aSrc + k0 + 4);
            *(float4*)(&areg[8])  = *(const float4*)(aSrc + k0 + 8);
            *(float4*)(&areg[12]) = *(const float4*)(aSrc + k0 + 12);
        }
#pragma unroll
        for (int i = 0; i < 16; ++i)
            breg[i] = bSrc[(size_t)(k0 + bk0 + i) * Ncols];
    };
    // Convert staged registers and store into LDS buffer `buf`.
    auto commit = [&](int buf) {
        if constexpr (sizeof(TA) != 2) {
            v8h a0, a1;
#pragma unroll
            for (int i = 0; i < 8; ++i) {
                a0[i] = (_Float16)areg[i];  a1[i] = (_Float16)areg[i + 8];
            }
            _Float16* da = &ldsA[buf][arow * LST + acol];
            *(v8h*)(da) = a0;  *(v8h*)(da + 8) = a1;
        }
        v8h b0, b1;
#pragma unroll
        for (int i = 0; i < 8; ++i) {
            b0[i] = (_Float16)breg[i];  b1[i] = (_Float16)breg[i + 8];
        }
        _Float16* db = &ldsB[buf][bcol * LST + bk0];   // transposed: [n][k]
        *(v8h*)(db) = b0;  *(v8h*)(db + 8) = b1;
    };

    fetch(0, 0);
    commit(0);
    if constexpr (sizeof(TA) == 2) async_wait_all();
    __syncthreads();

    for (int k0 = 0; k0 < Kd; k0 += BK) {
        const int cur = (k0 / BK) & 1, nxt = cur ^ 1;
        const bool hn = (k0 + BK) < Kd;
        if (hn) fetch(k0 + BK, nxt);   // globals/asyncs in flight across WMMAs

        v16h af[2], bf[4];
#pragma unroll
        for (int mi = 0; mi < 2; ++mi)
            af[mi] = load_frag(&ldsA[cur][(waveM * 32 + mi * 16 + lr) * LST + hi * 8]);
#pragma unroll
        for (int ni = 0; ni < 4; ++ni)
            bf[ni] = load_frag(&ldsB[cur][(waveN * 64 + ni * 16 + lr) * LST + hi * 8]);
#pragma unroll
        for (int mi = 0; mi < 2; ++mi)
#pragma unroll
            for (int ni = 0; ni < 4; ++ni)
                acc[mi][ni] = wmma_f16(af[mi], bf[ni], acc[mi][ni]);

        if (hn) commit(nxt);
        if constexpr (sizeof(TA) == 2) { if (hn) async_wait_all(); }
        __syncthreads();
    }

#pragma unroll
    for (int mi = 0; mi < 2; ++mi)
#pragma unroll
        for (int ni = 0; ni < 4; ++ni)
#pragma unroll
            for (int r = 0; r < 8; ++r) {
                int row = mBase + waveM * 32 + mi * 16 + hi * 8 + r;
                int col = nBase + waveN * 64 + ni * 16 + lr;
                C[(size_t)row * Ncols + col] = (TC)(acc[mi][ni][r] * alpha);
            }
}

// ======================================================================
// C[M,N] = alpha * A[M,Kd] * Bt[N,Kd]^T  (both K-contiguous f16)
// Fully async global->LDS (ASYNCcnt path), double-buffered.
// ======================================================================
__device__ __forceinline__ void gemm_nt(const _Float16* __restrict__ A,
                                        const _Float16* __restrict__ Bt,
                                        _Float16* __restrict__ C, int Kd, int Ncols,
                                        float alpha, int mBase, int nBase) {
    __shared__ __align__(16) _Float16 ldsA[2][BM * LST];
    __shared__ __align__(16) _Float16 ldsB[2][BM * LST];

    const int t    = threadIdx.x;
    const int lane = t & 31;
    const int wave = t >> 5;
    const int lr   = lane & 15;
    const int hi   = lane >> 4;
    const int waveM = wave & 3;
    const int waveN = wave >> 2;

    v8f acc[2][4];
#pragma unroll
    for (int mi = 0; mi < 2; ++mi)
#pragma unroll
        for (int ni = 0; ni < 4; ++ni) acc[mi][ni] = (v8f){0,0,0,0,0,0,0,0};

    const int arow = t >> 1;
    const int acol = (t & 1) * 16;
    const _Float16* aSrc = A  + (size_t)(mBase + arow) * Kd + acol;
    const _Float16* bSrc = Bt + (size_t)(nBase + arow) * Kd + acol;

    auto stage = [&](int k0, int buf) {
        _Float16* da = &ldsA[buf][arow * LST + acol];
        _Float16* db = &ldsB[buf][arow * LST + acol];
        cp_b128(aSrc + k0,     da);
        cp_b128(aSrc + k0 + 8, da + 8);
        cp_b128(bSrc + k0,     db);
        cp_b128(bSrc + k0 + 8, db + 8);
    };

    stage(0, 0);
    async_wait_all();
    __syncthreads();

    for (int k0 = 0; k0 < Kd; k0 += BK) {
        const int cur = (k0 / BK) & 1, nxt = cur ^ 1;
        const bool hn = (k0 + BK) < Kd;
        if (hn) stage(k0 + BK, nxt);   // async copies overlap the WMMAs

        v16h af[2], bf[4];
#pragma unroll
        for (int mi = 0; mi < 2; ++mi)
            af[mi] = load_frag(&ldsA[cur][(waveM * 32 + mi * 16 + lr) * LST + hi * 8]);
#pragma unroll
        for (int ni = 0; ni < 4; ++ni)
            bf[ni] = load_frag(&ldsB[cur][(waveN * 64 + ni * 16 + lr) * LST + hi * 8]);
#pragma unroll
        for (int mi = 0; mi < 2; ++mi)
#pragma unroll
            for (int ni = 0; ni < 4; ++ni)
                acc[mi][ni] = wmma_f16(af[mi], bf[ni], acc[mi][ni]);

        if (hn) async_wait_all();
        __syncthreads();
    }

#pragma unroll
    for (int mi = 0; mi < 2; ++mi)
#pragma unroll
        for (int ni = 0; ni < 4; ++ni)
#pragma unroll
            for (int r = 0; r < 8; ++r) {
                int row = mBase + waveM * 32 + mi * 16 + hi * 8 + r;
                int col = nBase + waveN * 64 + ni * 16 + lr;
                C[(size_t)row * Ncols + col] = (_Float16)(acc[mi][ni][r] * alpha);
            }
}

// ---------------- kernels ----------------

__global__ void __launch_bounds__(256)
qkv_kernel(const float* __restrict__ x,
           const float* __restrict__ Wq, const float* __restrict__ Wk,
           const float* __restrict__ Wv,
           _Float16* __restrict__ Q, _Float16* __restrict__ Kp, _Float16* __restrict__ Vp) {
    const float* W = (blockIdx.z == 0) ? Wq : (blockIdx.z == 1) ? Wk : Wv;
    _Float16*    O = (blockIdx.z == 0) ? Q  : (blockIdx.z == 1) ? Kp : Vp;
    gemm_bt<float, float, _Float16>(x, W, O, AD, AD, 1.0f,
                                    blockIdx.y * BM, blockIdx.x * BN);
}

__global__ void __launch_bounds__(256)
scores_kernel(const _Float16* __restrict__ Q, const _Float16* __restrict__ Kp,
              _Float16* __restrict__ S) {
    gemm_nt(Q, Kp, S, AD, AN, 0.03125f /* 1/sqrt(1024) */,
            blockIdx.y * BM, blockIdx.x * BN);
}

__global__ void __launch_bounds__(256)
softmax_kernel(_Float16* __restrict__ S) {
    __shared__ float red[8];
    const int row = blockIdx.x;
    const int t   = threadIdx.x;
    _Float16* p = S + (size_t)row * AN;

    float v[32];
    float m = -3.4e38f;
#pragma unroll
    for (int i = 0; i < 32; ++i) {
        v[i] = (float)p[t + 256 * i];
        m = fmaxf(m, v[i]);
    }
#pragma unroll
    for (int off = 16; off; off >>= 1) m = fmaxf(m, __shfl_xor(m, off, 32));
    if ((t & 31) == 0) red[t >> 5] = m;
    __syncthreads();
    m = red[0];
#pragma unroll
    for (int i = 1; i < 8; ++i) m = fmaxf(m, red[i]);
    __syncthreads();

    float s = 0.0f;
#pragma unroll
    for (int i = 0; i < 32; ++i) {
        v[i] = __expf(v[i] - m);
        s += v[i];
    }
#pragma unroll
    for (int off = 16; off; off >>= 1) s += __shfl_xor(s, off, 32);
    if ((t & 31) == 0) red[t >> 5] = s;
    __syncthreads();
    s = red[0];
#pragma unroll
    for (int i = 1; i < 8; ++i) s += red[i];
    const float inv = 1.0f / s;
#pragma unroll
    for (int i = 0; i < 32; ++i) p[t + 256 * i] = (_Float16)(v[i] * inv);
}

__global__ void __launch_bounds__(256)
out_kernel(const _Float16* __restrict__ P, const _Float16* __restrict__ Vp,
           float* __restrict__ O) {
    gemm_bt<_Float16, _Float16, float>(P, Vp, O, AN, AD, 1.0f,
                                       blockIdx.y * BM, blockIdx.x * BN);
}

extern "C" void kernel_launch(void* const* d_in, const int* in_sizes, int n_in,
                              void* d_out, int out_size, void* d_ws, size_t ws_size,
                              hipStream_t stream) {
    (void)in_sizes; (void)n_in; (void)out_size; (void)ws_size;
    const float* x  = (const float*)d_in[0];
    const float* Wq = (const float*)d_in[1];
    const float* Wk = (const float*)d_in[2];
    const float* Wv = (const float*)d_in[3];

    _Float16* Q  = (_Float16*)d_ws;
    _Float16* Kp = Q  + (size_t)AN * AD;
    _Float16* Vp = Kp + (size_t)AN * AD;
    _Float16* S  = Vp + (size_t)AN * AD;   // 8192*8192 f16; P overwrites in place

    dim3 blk(256);

    qkv_kernel<<<dim3(AD / BN, AN / BM, 3), blk, 0, stream>>>(x, Wq, Wk, Wv, Q, Kp, Vp);
    scores_kernel<<<dim3(AN / BN, AN / BM), blk, 0, stream>>>(Q, Kp, S);
    softmax_kernel<<<dim3(AN), blk, 0, stream>>>(S);
    out_kernel<<<dim3(AD / BN, AN / BM), blk, 0, stream>>>(S, Vp, (float*)d_out);
}
